// GCN_15530601742784
// MI455X (gfx1250) — matrix operand order repaired and checked by reading, not compiled, not run on hardware
//
#include <hip/hip_runtime.h>
#include <hip/hip_bf16.h>

// Problem constants (match reference)
#define NNODES 100000
#define NPAD   100032              // 1563 * 64 (GEMM row-tile padding)
#define NEDGES 1600000
#define HID    128
#define NGRAPH 64
#define H2D    64
#define H3D    32
#define OUTD   2
#define EPSV   1e-5f

typedef __bf16 v16bf __attribute__((ext_vector_type(16)));
typedef float  v8f   __attribute__((ext_vector_type(8)));
typedef int    v4i   __attribute__((vector_size(16)));

#if defined(__gfx1250__) && __has_builtin(__builtin_amdgcn_global_load_async_to_lds_b128)
#define ASYNC_STAGE 1
#else
#define ASYNC_STAGE 0
#endif

#if ASYNC_STAGE
// builtin signature (probe-derived): (v4i __device__* src, v4i __shared__* dst, imm off, imm cpol)
#define GAS1(p) ((__attribute__((address_space(1))) v4i*)(p))
#define LAS3(p) ((__attribute__((address_space(3))) v4i*)(p))
__device__ __forceinline__ void async_wait0() {
#if __has_builtin(__builtin_amdgcn_s_wait_asynccnt)
  __builtin_amdgcn_s_wait_asynccnt(0);
#else
  asm volatile("s_wait_asynccnt 0x0" ::: "memory");
#endif
}
#endif

__device__ __forceinline__ unsigned short f2bf(float f) {
  unsigned int u = __float_as_uint(f);
  u += 0x7FFFu + ((u >> 16) & 1u);   // round-to-nearest-even
  return (unsigned short)(u >> 16);
}

// ---------------- edge prep ----------------
__global__ void edge_cvt(const long long* __restrict__ ei,
                         int* __restrict__ s32, int* __restrict__ d32) {
  int e = blockIdx.x * blockDim.x + threadIdx.x;
  if (e < NEDGES) { s32[e] = (int)ei[e]; d32[e] = (int)ei[NEDGES + e]; }
}

// ---------------- degree / norm ----------------
__global__ void deg_init(float* deg) {
  int i = blockIdx.x * blockDim.x + threadIdx.x;
  if (i < NNODES) deg[i] = 1.0f;     // self-loop
}
__global__ void deg_edges(const int* __restrict__ d32, float* deg) {
  int e = blockIdx.x * blockDim.x + threadIdx.x;
  if (e < NEDGES) atomicAdd(&deg[d32[e]], 1.0f);
}
__global__ void deg_to_dis(float* deg) {
  int i = blockIdx.x * blockDim.x + threadIdx.x;
  if (i < NNODES) deg[i] = rsqrtf(deg[i]);
}
__global__ void edge_coef(const int* __restrict__ s32, const int* __restrict__ d32,
                          const float* __restrict__ dis, float* __restrict__ coef) {
  int e = blockIdx.x * blockDim.x + threadIdx.x;
  if (e < NEDGES) coef[e] = dis[s32[e]] * dis[d32[e]];
}

// ---------------- operand packing ----------------
// Wpack element i (i = ((kk*8 + ct)*32 + lane)*16 + e) holds
// bf16( W[(kk*32 + e + (lane>=16?16:0)) * 128 + ct*16 + (lane&15)] )
// so a B fragment is 32 contiguous bytes per lane.
__global__ void pack_w(const float* __restrict__ W, unsigned short* __restrict__ Wp) {
  int i = blockIdx.x * blockDim.x + threadIdx.x;
  if (i >= 4 * 8 * 32 * 16) return;
  int e = i & 15, lane = (i >> 4) & 31, ct = (i >> 9) & 7, kk = i >> 12;
  int K = kk * 32 + e + ((lane >> 4) << 4);
  int n = ct * 16 + (lane & 15);
  Wp[i] = f2bf(W[K * 128 + n]);
}

// x (f32) -> bf16 row-major, zero padded rows
__global__ void to_bf16_pad(const float* __restrict__ X, unsigned short* __restrict__ Xb) {
  int i = blockIdx.x * blockDim.x + threadIdx.x;
  if (i < NNODES * 128) Xb[i] = f2bf(X[i]);
  else if (i < NPAD * 128) Xb[i] = 0;
}

// ---------------- WMMA GEMM: out[NPAD,128] = A[NPAD,128](bf16) @ W[128,128] ----------------
// 256 threads = 8 waves; block does 64 rows; wave -> 16 rows x 64 cols (4 tiles)
__global__ void gemm_wmma(const unsigned short* __restrict__ Abf,
                          const unsigned short* __restrict__ Wpack,
                          float* __restrict__ out) {
  __shared__ unsigned short sA[64 * 128];   // 16 KB, row-major bf16
  __shared__ unsigned short sW[16384];      // 32 KB, fragment-packed bf16
  const int tid     = threadIdx.x;
  const int rowBase = blockIdx.x * 64;

  const uint4* gA = (const uint4*)(Abf + (size_t)rowBase * 128);
  const uint4* gW = (const uint4*)Wpack;
  uint4*       lA = (uint4*)sA;
  uint4*       lW = (uint4*)sW;
#if ASYNC_STAGE
  // async global -> LDS staging (ASYNCcnt path), 16B per issue
  for (int i = tid; i < (64 * 128) / 8; i += 256)
    __builtin_amdgcn_global_load_async_to_lds_b128(GAS1(gA + i), LAS3(lA + i), 0, 0);
  for (int i = tid; i < 16384 / 8; i += 256)
    __builtin_amdgcn_global_load_async_to_lds_b128(GAS1(gW + i), LAS3(lW + i), 0, 0);
  async_wait0();
  __syncthreads();
#else
  for (int i = tid; i < (64 * 128) / 8; i += 256) lA[i] = gA[i];
  for (int i = tid; i < 16384 / 8; i += 256)      lW[i] = gW[i];
  __syncthreads();
#endif

  const int wave   = tid >> 5;
  const int lane   = tid & 31;
  const int laneHi = lane >> 4;
  const int m16    = lane & 15;
  const int rt     = wave >> 1;    // row tile 0..3
  const int ch     = wave & 1;     // col half: tiles ch*4 .. ch*4+3
  const v16bf* sWv = (const v16bf*)sW;

  v8f acc[4] = {};
  #pragma unroll
  for (int kk = 0; kk < 4; ++kk) {
    // A fragment: one contiguous 32B run per lane (k = 16*laneHi + s, matches Wpack)
    union { struct { uint4 lo, hi; } u; v16bf v; } av;
    const uint4* ap = (const uint4*)&sA[(rt * 16 + m16) * 128 + kk * 32 + laneHi * 16];
    av.u.lo = ap[0];
    av.u.hi = ap[1];
    v16bf a = av.v;
    #pragma unroll
    for (int t = 0; t < 4; ++t) {
      const int ct = ch * 4 + t;
      v16bf b = sWv[(kk * 8 + ct) * 32 + lane];
      acc[t] = __builtin_amdgcn_wmma_f32_16x16x32_bf16(false, a, false, b, (short)0,
                                                       acc[t], false, false);
    }
  }

  // C/D layout: acc[t][r] -> row = r + laneHi*8, col = (ch*4+t)*16 + lane%16
  #pragma unroll
  for (int t = 0; t < 4; ++t) {
    float* orow = out + (size_t)(rowBase + rt * 16 + laneHi * 8) * 128 + (ch * 4 + t) * 16 + m16;
    #pragma unroll
    for (int r = 0; r < 8; ++r) orow[(size_t)r * 128] = acc[t][r];
  }
}

// ---------------- aggregation ----------------
__global__ void self_init(const float* __restrict__ xw, const float* __restrict__ dis,
                          float* __restrict__ agg) {
  int n = blockIdx.x, c = threadIdx.x;
  float di = dis[n];
  agg[(size_t)n * 128 + c] = xw[(size_t)n * 128 + c] * di * di;
}

// one wave per edge: 32 lanes x 4 channels
__global__ void edge_agg(const int* __restrict__ s32, const int* __restrict__ d32,
                         const float* __restrict__ coef, const float* __restrict__ xw,
                         float* __restrict__ agg) {
  int t    = blockIdx.x * blockDim.x + threadIdx.x;
  int edge = t >> 5;
  int q    = (t & 31) * 4;
  if (edge >= NEDGES) return;
  int s = s32[edge];
  int d = d32[edge];
  float cf = coef[edge];
  float4 v = *(const float4*)(xw + (size_t)s * 128 + q);
  float* ap = agg + (size_t)d * 128 + q;
  atomicAdd(ap + 0, v.x * cf);
  atomicAdd(ap + 1, v.y * cf);
  atomicAdd(ap + 2, v.z * cf);
  atomicAdd(ap + 3, v.w * cf);
}

// BN + ReLU; writes f32 (for pooling / in-place h) and bf16 copy (next GEMM operand)
__global__ void bn_relu(float* __restrict__ agg, unsigned short* __restrict__ hbf,
                        const float* __restrict__ b, const float* __restrict__ g,
                        const float* __restrict__ be, const float* __restrict__ m,
                        const float* __restrict__ v) {
  int n = blockIdx.x, c = threadIdx.x;
  size_t idx = (size_t)n * 128 + c;
  float x = agg[idx] + b[c];
  x = (x - m[c]) * rsqrtf(v[c] + EPSV) * g[c] + be[c];
  x = fmaxf(x, 0.0f);
  agg[idx] = x;
  hbf[idx] = f2bf(x);
}

// ---------------- pooling + MLP ----------------
__global__ void pool_init(float* p) {
  int i = blockIdx.x * blockDim.x + threadIdx.x;
  if (i < NGRAPH * 128 + NGRAPH) p[i] = 0.0f;
}
__global__ void pool_accum(const float* __restrict__ h, const long long* __restrict__ batch,
                           float* __restrict__ pooled, float* __restrict__ cnt) {
  int n = blockIdx.x, c = threadIdx.x;
  int gi = (int)batch[n];
  atomicAdd(&pooled[gi * 128 + c], h[(size_t)n * 128 + c]);
  if (c == 0) atomicAdd(&cnt[gi], 1.0f);
}

__global__ void mlp_head(const float* __restrict__ pooled, const float* __restrict__ cnt,
                         const float* __restrict__ Wm1, const float* __restrict__ bm1,
                         const float* __restrict__ Wm2, const float* __restrict__ bm2,
                         const float* __restrict__ Wm3, const float* __restrict__ bm3,
                         const float* __restrict__ Wo,  const float* __restrict__ bo,
                         float* __restrict__ out) {
  __shared__ float zA[NGRAPH * 128];
  __shared__ float zB[NGRAPH * 128];
  int tid = threadIdx.x;
  for (int i = tid; i < NGRAPH * 128; i += 256) {
    int r = i >> 7;
    zA[i] = pooled[i] / fmaxf(cnt[r], 1.0f);
  }
  __syncthreads();
  for (int i = tid; i < NGRAPH * 128; i += 256) {         // 128 -> 128
    int r = i >> 7, c = i & 127;
    float acc = bm1[c];
    for (int k = 0; k < 128; ++k) acc += zA[r * 128 + k] * Wm1[k * 128 + c];
    zB[i] = fmaxf(acc, 0.0f);
  }
  __syncthreads();
  for (int i = tid; i < NGRAPH * H2D; i += 256) {         // 128 -> 64
    int r = i >> 6, c = i & 63;
    float acc = bm2[c];
    for (int k = 0; k < 128; ++k) acc += zB[r * 128 + k] * Wm2[k * H2D + c];
    zA[r * H2D + c] = fmaxf(acc, 0.0f);
  }
  __syncthreads();
  for (int i = tid; i < NGRAPH * H3D; i += 256) {         // 64 -> 32
    int r = i >> 5, c = i & 31;
    float acc = bm3[c];
    for (int k = 0; k < H2D; ++k) acc += zA[r * H2D + k] * Wm3[k * H3D + c];
    zB[r * H3D + c] = fmaxf(acc, 0.0f);
  }
  __syncthreads();
  for (int i = tid; i < NGRAPH * OUTD; i += 256) {        // 32 -> 2
    int r = i >> 1, c = i & 1;
    float acc = bo[c];
    for (int k = 0; k < H3D; ++k) acc += zB[r * H3D + k] * Wo[k * OUTD + c];
    out[i] = acc;
  }
}

static inline size_t align256(size_t x) { return (x + 255) & ~(size_t)255; }

extern "C" void kernel_launch(void* const* d_in, const int* in_sizes, int n_in,
                              void* d_out, int out_size, void* d_ws, size_t ws_size,
                              hipStream_t stream) {
  const float*     x     = (const float*)d_in[0];
  const long long* ei    = (const long long*)d_in[1];
  const long long* batch = (const long long*)d_in[2];
  const float* W[3]  = { (const float*)d_in[3],  (const float*)d_in[9],  (const float*)d_in[15] };
  const float* bb[3] = { (const float*)d_in[4],  (const float*)d_in[10], (const float*)d_in[16] };
  const float* gg[3] = { (const float*)d_in[5],  (const float*)d_in[11], (const float*)d_in[17] };
  const float* be[3] = { (const float*)d_in[6],  (const float*)d_in[12], (const float*)d_in[18] };
  const float* mm[3] = { (const float*)d_in[7],  (const float*)d_in[13], (const float*)d_in[19] };
  const float* vv[3] = { (const float*)d_in[8],  (const float*)d_in[14], (const float*)d_in[20] };
  const float* Wm1 = (const float*)d_in[21]; const float* bm1 = (const float*)d_in[22];
  const float* Wm2 = (const float*)d_in[23]; const float* bm2 = (const float*)d_in[24];
  const float* Wm3 = (const float*)d_in[25]; const float* bm3 = (const float*)d_in[26];
  const float* Wo  = (const float*)d_in[27]; const float* bo  = (const float*)d_in[28];
  float* out = (float*)d_out;

  // workspace layout
  char*  ws  = (char*)d_ws;
  size_t off = 0;
  float* dis = (float*)(ws + off); off += align256((size_t)NNODES * 4);
  float* xw  = (float*)(ws + off); off += align256((size_t)NPAD * 128 * 4);
  float* agg = (float*)(ws + off); off += align256((size_t)NNODES * 128 * 4);
  unsigned short* abf = (unsigned short*)(ws + off); off += align256((size_t)NPAD * 128 * 2);
  unsigned short* Wpk = (unsigned short*)(ws + off); off += align256((size_t)3 * 16384 * 2);
  int*   s32  = (int*)(ws + off);  off += align256((size_t)NEDGES * 4);
  int*   d32  = (int*)(ws + off);  off += align256((size_t)NEDGES * 4);
  float* coef = (float*)(ws + off); off += align256((size_t)NEDGES * 4);
  float* pooled = (float*)(ws + off); off += align256((size_t)NGRAPH * 128 * 4);
  float* cnt    = (float*)(ws + off); off += align256((size_t)NGRAPH * 4);
  (void)ws_size; (void)in_sizes; (void)n_in; (void)out_size;

  // prep: int32 edges, degrees, dis = 1/sqrt(deg), per-edge coef
  edge_cvt<<<(NEDGES + 255) / 256, 256, 0, stream>>>(ei, s32, d32);
  deg_init<<<(NNODES + 255) / 256, 256, 0, stream>>>(dis);
  deg_edges<<<(NEDGES + 255) / 256, 256, 0, stream>>>(d32, dis);
  deg_to_dis<<<(NNODES + 255) / 256, 256, 0, stream>>>(dis);
  edge_coef<<<(NEDGES + 255) / 256, 256, 0, stream>>>(s32, d32, dis, coef);

  // prep: fragment-packed bf16 weights, bf16 node features (padded)
  for (int l = 0; l < 3; ++l)
    pack_w<<<16384 / 256, 256, 0, stream>>>(W[l], Wpk + (size_t)l * 16384);
  to_bf16_pad<<<(NPAD * 128) / 256, 256, 0, stream>>>(x, abf);

  pool_init<<<(NGRAPH * 128 + NGRAPH + 255) / 256, 256, 0, stream>>>(pooled);

  for (int l = 0; l < 3; ++l) {
    gemm_wmma<<<NPAD / 64, 256, 0, stream>>>(abf, Wpk + (size_t)l * 16384, xw);
    self_init<<<NNODES, 128, 0, stream>>>(xw, dis, agg);
    edge_agg<<<(NEDGES * 32) / 256, 256, 0, stream>>>(s32, d32, coef, xw, agg);
    bn_relu<<<NNODES, 128, 0, stream>>>(agg, abf, bb[l], gg[l], be[l], mm[l], vv[l]);
  }

  pool_accum<<<NNODES, 128, 0, stream>>>(agg, batch, pooled, cnt);
  mlp_head<<<1, 256, 0, stream>>>(pooled, cnt, Wm1, bm1, Wm2, bm2, Wm3, bm3, Wo, bo, out);
}
